// TpjBrain_56341380989375
// MI455X (gfx1250) — matrix operand order repaired and checked by reading, not compiled, not run on hardware
//
#include <hip/hip_runtime.h>

// ---------------------------------------------------------------------------
// GPT-2 small forward (B=2, T=1024, E=768, H=12, L=8, V=50257) for gfx1250.
// GEMMs: V_WMMA_F32_16X16X32_BF16, 128x128 block tile, 8 waves (4x2),
// 32x64 strip per wave (8 WMMAs/K-step), double-buffered LDS with
// GLOBAL_LOAD_ASYNC_TO_LDS_B128 overlapping compute.
// ---------------------------------------------------------------------------

typedef __attribute__((ext_vector_type(16))) __bf16 v16bf;
typedef __attribute__((ext_vector_type(8)))  __bf16 v8bf;
typedef __attribute__((ext_vector_type(8)))  float  v8f;

union frag16 { v16bf v; v8bf h[2]; };

#define GPT_B 2
#define GPT_T 1024
#define GPT_E 768
#define GPT_H 12
#define GPT_L 8
#define GPT_V 50257
#define GPT_M (GPT_B * GPT_T)   // 2048 rows of activations

__device__ __forceinline__ float gelu_f(float x) {
    float x3 = x * x * x;
    return 0.5f * x * (1.0f + tanhf(0.7978845608028654f * (x + 0.044715f * x3)));
}

// CDNA5 async copy: 16B per lane, global (SADDR + 32-bit voffset) -> LDS.
__device__ __forceinline__ void async_copy16(unsigned lds_off, unsigned gl_off,
                                             const void* sbase) {
    asm volatile("global_load_async_to_lds_b128 %0, %1, %2"
                 :: "v"(lds_off), "v"(gl_off), "s"(sbase)
                 : "memory");
}
__device__ __forceinline__ void wait_async0() {
    asm volatile("s_wait_asynccnt 0x0" ::: "memory");
}
__device__ __forceinline__ void compiler_mem_barrier() {
    asm volatile("" ::: "memory");
}

// ---------------------------------------------------------------------------
// Embedding (fp32 residual stream)
// ---------------------------------------------------------------------------
__global__ __launch_bounds__(256) void embed_kernel(const int* __restrict__ idx,
                                                    const float* __restrict__ wte,
                                                    const float* __restrict__ wpe,
                                                    float* __restrict__ x) {
    int row = blockIdx.x;
    int t   = row & (GPT_T - 1);
    int tok = idx[row];
    const float* we = wte + (size_t)tok * GPT_E;
    const float* pe = wpe + (size_t)t * GPT_E;
    float* xo = x + (size_t)row * GPT_E;
    for (int i = threadIdx.x; i < GPT_E; i += 256)
        xo[i] = we[i] + pe[i];
}

// ---------------------------------------------------------------------------
// LayerNorm: fp32 in, bf16 out
// ---------------------------------------------------------------------------
__global__ __launch_bounds__(256) void ln_kernel(const float* __restrict__ x,
                                                 const float* __restrict__ g,
                                                 const float* __restrict__ b,
                                                 __bf16* __restrict__ out) {
    __shared__ float red[256];
    int row = blockIdx.x;
    int tid = threadIdx.x;
    const float* xr = x + (size_t)row * GPT_E;

    float s = 0.f;
    for (int i = tid; i < GPT_E; i += 256) s += xr[i];
    red[tid] = s; __syncthreads();
    for (int st = 128; st > 0; st >>= 1) {
        if (tid < st) red[tid] += red[tid + st];
        __syncthreads();
    }
    float mean = red[0] * (1.0f / GPT_E);
    __syncthreads();

    float vs = 0.f;
    for (int i = tid; i < GPT_E; i += 256) { float d = xr[i] - mean; vs += d * d; }
    red[tid] = vs; __syncthreads();
    for (int st = 128; st > 0; st >>= 1) {
        if (tid < st) red[tid] += red[tid + st];
        __syncthreads();
    }
    float inv = rsqrtf(red[0] * (1.0f / GPT_E) + 1e-5f);

    __bf16* orow = out + (size_t)row * GPT_E;
    for (int i = tid; i < GPT_E; i += 256)
        orow[i] = (__bf16)((xr[i] - mean) * inv * g[i] + b[i]);
}

// ---------------------------------------------------------------------------
// Transpose V section of qkv into vt[b][h][d][t] (bf16)
// ---------------------------------------------------------------------------
__global__ __launch_bounds__(256) void vtrans_kernel(const __bf16* __restrict__ qkv,
                                                     __bf16* __restrict__ vt) {
    int t0 = blockIdx.x * 64;
    int h  = blockIdx.y, b = blockIdx.z;
    int tl = threadIdx.x & 63;
    int dl = threadIdx.x >> 6;
    const __bf16* src = qkv + ((size_t)(b * GPT_T + t0 + tl)) * (3 * GPT_E)
                        + 2 * GPT_E + h * 64 + dl * 16;
    __bf16* dst = vt + ((size_t)(b * GPT_H + h)) * 64 * GPT_T + t0 + tl;
    v8bf a = ((const v8bf*)src)[0];
    v8bf c = ((const v8bf*)src)[1];
    #pragma unroll
    for (int j = 0; j < 8; ++j) {
        dst[(size_t)(dl * 16 + j) * GPT_T]     = a[j];
        dst[(size_t)(dl * 16 + 8 + j) * GPT_T] = c[j];
    }
}

// ---------------------------------------------------------------------------
// GEMM (templated shape): C[2048,N] = act(A[2048,K](bf16) @ W[K,N](f32) + bias)
// (+ residual f32). Block tile 128x128, 8 waves (4 M x 2 N), wave strip 32x64.
// Double-buffered LDS: A via async copy, W via register-staged global loads.
// ---------------------------------------------------------------------------
#define TM 128
#define TN 128
#define KS 32
#define WKP (KS + 8)          // 40 elements = 80B pitch (16B aligned)
#define AS_BYTES (TM * KS * 2)

template <int N, int K, bool NGUARD>
__global__ __launch_bounds__(256) void gemm_wmma_kernel(
    const __bf16* __restrict__ A, const float* __restrict__ W,
    const float* __restrict__ bias, const float* __restrict__ res,
    float* __restrict__ Cf, __bf16* __restrict__ Ch, int act) {

    __shared__ __align__(16) __bf16 As[2][TM][KS];    // 2 x 8 KB
    __shared__ __align__(16) __bf16 Wst[2][TN][WKP];  // 2 x 10 KB

    const int tid  = threadIdx.x;
    const int wave = tid >> 5;
    const int lane = tid & 31;
    const int rlo  = lane & 15;
    const int hi   = lane >> 4;
    const int mw   = wave >> 1;          // 0..3 : 32-row strip
    const int nw   = wave & 1;           // 0..1 : 64-col strip

    const int bm = blockIdx.y * TM;
    const int bn = blockIdx.x * TN;

    // staging coordinates
    const int wn  = tid & 127;           // W: one column per thread pair
    const int wkb = (tid >> 7) * 16;     // 16 k values per thread

    const unsigned as_base = (unsigned)(size_t)&As[0][0][0];

    const int gn = bn + wn;
    const bool nval = !NGUARD || (gn < N);
    const float* wcol = W + (NGUARD ? (nval ? gn : 0) : gn);

    v8bf wreg[2];
    auto load_w = [&](int k0) {
        const float* wp = wcol + (size_t)(k0 + wkb) * N;
        #pragma unroll
        for (int j = 0; j < 16; ++j) {
            float t = wp[(size_t)j * N];
            wreg[j >> 3][j & 7] = nval ? (__bf16)t : (__bf16)0.0f;
        }
    };
    auto store_w = [&](int buf) {
        *(v8bf*)&Wst[buf][wn][wkb]     = wreg[0];
        *(v8bf*)&Wst[buf][wn][wkb + 8] = wreg[1];
    };
    auto issue_a = [&](int k0, int buf) {
        #pragma unroll
        for (int i = 0; i < 2; ++i) {
            int ch  = tid + i * 256;     // 0..511 chunks of 16B
            int row = ch >> 2, c = ch & 3;
            async_copy16(as_base + buf * AS_BYTES + row * (KS * 2) + c * 16,
                         (unsigned)(((bm + row) * K + k0) * 2) + c * 16,
                         A);
        }
    };

    v8f acc[2][4] = {};

    // prologue: fill buffer 0
    issue_a(0, 0);
    load_w(0);
    store_w(0);
    wait_async0();
    __syncthreads();

    for (int k0 = 0; k0 < K; k0 += KS) {
        const int  cb   = (k0 / KS) & 1;
        const bool more = (k0 + KS) < K;

        // start next tile's transfers before computing this one
        if (more) {
            issue_a(k0 + KS, 1 - cb);
            load_w(k0 + KS);
        }

        // ---- fragments from current buffer ----
        frag16 af[2];
        #pragma unroll
        for (int g = 0; g < 2; ++g) {
            const __bf16* ap = &As[cb][mw * 32 + g * 16 + rlo][0];
            af[g].h[0] = *(const v8bf*)&ap[hi * 8];
            af[g].h[1] = *(const v8bf*)&ap[16 + hi * 8];
        }
        frag16 bf[4];
        #pragma unroll
        for (int f = 0; f < 4; ++f) {
            const __bf16* bp = &Wst[cb][nw * 64 + f * 16 + rlo][hi * 8];
            bf[f].h[0] = *(const v8bf*)&bp[0];
            bf[f].h[1] = *(const v8bf*)&bp[16];
        }
        #pragma unroll
        for (int g = 0; g < 2; ++g)
            #pragma unroll
            for (int f = 0; f < 4; ++f)
                acc[g][f] = __builtin_amdgcn_wmma_f32_16x16x32_bf16(
                    false, af[g].v, false, bf[f].v, (short)0, acc[g][f], false, false);

        if (more) {
            store_w(1 - cb);
            wait_async0();
        }
        __syncthreads();
    }

    // ---- epilogue: bias, activation, residual, store (fp32 or bf16) ----
    #pragma unroll
    for (int g = 0; g < 2; ++g) {
        int mbase = bm + mw * 32 + g * 16 + hi * 8;
        #pragma unroll
        for (int f = 0; f < 4; ++f) {
            int n = bn + nw * 64 + f * 16 + rlo;
            if (NGUARD && n >= N) continue;
            float bv = bias ? bias[n] : 0.0f;
            #pragma unroll
            for (int r = 0; r < 8; ++r) {
                float v = acc[g][f][r] + bv;
                if (act) v = gelu_f(v);
                size_t off = (size_t)(mbase + r) * N + n;
                if (res) v += res[off];
                if (Cf) Cf[off] = v;
                else    Ch[off] = (__bf16)v;
            }
        }
    }
}

// ---------------------------------------------------------------------------
// Flash attention: 4 independent waves per block, each owns one (b,h,16-row
// q tile) and a private LDS slice. 32 keys per iteration, K/V async-copied.
// ---------------------------------------------------------------------------
__global__ __launch_bounds__(128) void attn_kernel(const __bf16* __restrict__ qkv,
                                                   const __bf16* __restrict__ vt,
                                                   __bf16* __restrict__ y) {
    const int wave = threadIdx.x >> 5;
    const int lane = threadIdx.x & 31;
    const int rlo  = lane & 15;
    const int hi   = lane >> 4;
    const int qt   = blockIdx.x * 4 + wave;
    const int h    = blockIdx.y;
    const int b    = blockIdx.z;
    const int m0   = qt * 16;
    const int ld   = 3 * GPT_E;

    const __bf16* qglob = qkv + (size_t)b * GPT_T * ld + (size_t)h * 64;
    const __bf16* kglob = qglob + GPT_E;
    const __bf16* vtg   = vt + ((size_t)(b * GPT_H + h)) * 64 * GPT_T;

    __shared__ __align__(16) __bf16 Kls[4][32][72];  // keys x d, pitch 144B
    __shared__ __align__(16) __bf16 Vls[4][64][40];  // d x keys, pitch 80B
    __shared__ __align__(16) __bf16 Pls[4][16][40];  // rows x keys, pitch 80B
    const unsigned kls_base = (unsigned)(size_t)&Kls[wave][0][0];
    const unsigned vls_base = (unsigned)(size_t)&Vls[wave][0][0];
    __bf16 (*Kw)[72] = Kls[wave];
    __bf16 (*Vw)[40] = Vls[wave];
    __bf16 (*Pw)[40] = Pls[wave];

    v16bf qf[2];
    {
        const __bf16* qrow = qglob + (size_t)(m0 + rlo) * ld;
        #pragma unroll
        for (int f = 0; f < 2; ++f) {
            frag16 q;
            q.h[0] = *(const v8bf*)&qrow[f * 32 + hi * 8];
            q.h[1] = *(const v8bf*)&qrow[f * 32 + 16 + hi * 8];
            qf[f] = q.v;
        }
    }

    v8f o[4] = {};
    float mrun[8], lrun[8];
    #pragma unroll
    for (int r = 0; r < 8; ++r) { mrun[r] = -3.0e38f; lrun[r] = 0.0f; }

    const int nb = (m0 + 47) / 32;
    for (int kt = 0; kt < nb; ++kt) {
        #pragma unroll
        for (int i = 0; i < 8; ++i) {
            int ch  = lane + i * 32;      // K: 256 chunks of 16B
            int row = ch >> 3, c = ch & 7;
            async_copy16(kls_base + row * 144 + c * 16,
                         (unsigned)(((kt * 32 + row) * ld) * 2) + c * 16,
                         kglob);
        }
        #pragma unroll
        for (int i = 0; i < 8; ++i) {
            int ch  = lane + i * 32;      // V: 256 chunks of 16B
            int row = ch >> 2, c = ch & 3;
            async_copy16(vls_base + row * 80 + c * 16,
                         (unsigned)((row * GPT_T + kt * 32) * 2) + c * 16,
                         vtg);
        }
        wait_async0();

        v8f s0 = {}, s1 = {};
        #pragma unroll
        for (int f = 0; f < 2; ++f) {
            frag16 kf;
            const __bf16* kp = &Kw[rlo][hi * 8 + f * 32];
            kf.h[0] = *(const v8bf*)&kp[0];
            kf.h[1] = *(const v8bf*)&kp[16];
            s0 = __builtin_amdgcn_wmma_f32_16x16x32_bf16(
                false, qf[f], false, kf.v, (short)0, s0, false, false);
        }
        #pragma unroll
        for (int f = 0; f < 2; ++f) {
            frag16 kf;
            const __bf16* kp = &Kw[16 + rlo][hi * 8 + f * 32];
            kf.h[0] = *(const v8bf*)&kp[0];
            kf.h[1] = *(const v8bf*)&kp[16];
            s1 = __builtin_amdgcn_wmma_f32_16x16x32_bf16(
                false, qf[f], false, kf.v, (short)0, s1, false, false);
        }

        int ki0 = kt * 32 + rlo;
        int ki1 = ki0 + 16;
        #pragma unroll
        for (int r = 0; r < 8; ++r) {
            int qi = m0 + hi * 8 + r;
            float v0 = s0[r] * 0.125f;
            float v1 = s1[r] * 0.125f;
            if (ki0 > qi) v0 = -1.0e30f;
            if (ki1 > qi) v1 = -1.0e30f;
            float mx = fmaxf(v0, v1);
            #pragma unroll
            for (int off = 1; off < 16; off <<= 1)
                mx = fmaxf(mx, __shfl_xor(mx, off, 16));
            float nm = fmaxf(mrun[r], mx);
            float p0 = expf(v0 - nm);
            float p1 = expf(v1 - nm);
            float ps = p0 + p1;
            #pragma unroll
            for (int off = 1; off < 16; off <<= 1)
                ps += __shfl_xor(ps, off, 16);
            float alpha = expf(mrun[r] - nm);
            lrun[r] = lrun[r] * alpha + ps;
            mrun[r] = nm;
            #pragma unroll
            for (int f = 0; f < 4; ++f) o[f][r] *= alpha;
            Pw[hi * 8 + r][rlo]      = (__bf16)p0;
            Pw[hi * 8 + r][16 + rlo] = (__bf16)p1;
        }
        compiler_mem_barrier();    // per-wave DS ops execute in order

        frag16 pf;
        pf.h[0] = *(const v8bf*)&Pw[rlo][hi * 8];
        pf.h[1] = *(const v8bf*)&Pw[rlo][16 + hi * 8];

        #pragma unroll
        for (int f = 0; f < 4; ++f) {
            frag16 vf;
            const __bf16* vp = &Vw[f * 16 + rlo][hi * 8];
            vf.h[0] = *(const v8bf*)&vp[0];
            vf.h[1] = *(const v8bf*)&vp[16];
            o[f] = __builtin_amdgcn_wmma_f32_16x16x32_bf16(
                false, pf.v, false, vf.v, (short)0, o[f], false, false);
        }
    }

    #pragma unroll
    for (int f = 0; f < 4; ++f) {
        #pragma unroll
        for (int r = 0; r < 8; ++r) {
            size_t row = (size_t)b * GPT_T + m0 + hi * 8 + r;
            y[row * GPT_E + h * 64 + f * 16 + rlo] = (__bf16)(o[f][r] / lrun[r]);
        }
    }
}

// ---------------------------------------------------------------------------
// Host orchestration
// ---------------------------------------------------------------------------
extern "C" void kernel_launch(void* const* d_in, const int* in_sizes, int n_in,
                              void* d_out, int out_size, void* d_ws, size_t ws_size,
                              hipStream_t stream) {
    (void)in_sizes; (void)n_in; (void)out_size; (void)ws_size;

    const int*   idx    = (const int*)  d_in[0];
    const float* wte    = (const float*)d_in[1];
    const float* wpe    = (const float*)d_in[2];
    const float* ln1_g  = (const float*)d_in[3];
    const float* ln1_b  = (const float*)d_in[4];
    const float* attn_w = (const float*)d_in[5];
    const float* attn_b = (const float*)d_in[6];
    const float* proj_w = (const float*)d_in[7];
    const float* proj_b = (const float*)d_in[8];
    const float* ln2_g  = (const float*)d_in[9];
    const float* ln2_b  = (const float*)d_in[10];
    const float* mlp_w1 = (const float*)d_in[11];
    const float* mlp_b1 = (const float*)d_in[12];
    const float* mlp_w2 = (const float*)d_in[13];
    const float* mlp_b2 = (const float*)d_in[14];
    const float* lnf_g  = (const float*)d_in[15];
    const float* lnf_b  = (const float*)d_in[16];
    const float* head_w = (const float*)d_in[17];
    float* out = (float*)d_out;

    char* w0 = (char*)d_ws;
    float*  xbuf = (float*)w0;                                              // [2048,768] f32
    __bf16* hbuf = (__bf16*)(w0 + (size_t)GPT_M * GPT_E * 4);               // [2048,768]
    __bf16* qkvb = (__bf16*)((char*)hbuf + (size_t)GPT_M * GPT_E * 2);      // [2048,2304]
    __bf16* ybuf = (__bf16*)((char*)qkvb + (size_t)GPT_M * 3 * GPT_E * 2);  // [2048,768]
    __bf16* mlph = (__bf16*)((char*)ybuf + (size_t)GPT_M * GPT_E * 2);      // [2048,3072]
    __bf16* vtb  = (__bf16*)((char*)mlph + (size_t)GPT_M * 4 * GPT_E * 2);  // [B,H,64,T]

    dim3 blk(256);
    dim3 gM(GPT_M);

    embed_kernel<<<gM, blk, 0, stream>>>(idx, wte, wpe, xbuf);

    for (int l = 0; l < GPT_L; ++l) {
        ln_kernel<<<gM, blk, 0, stream>>>(xbuf, ln1_g + l * GPT_E, ln1_b + l * GPT_E, hbuf);
        gemm_wmma_kernel<3 * GPT_E, GPT_E, false>
            <<<dim3(3 * GPT_E / TN, GPT_M / TM), blk, 0, stream>>>(
            hbuf, attn_w + (size_t)l * GPT_E * 3 * GPT_E, attn_b + (size_t)l * 3 * GPT_E,
            nullptr, nullptr, qkvb, 0);
        vtrans_kernel<<<dim3(GPT_T / 64, GPT_H, GPT_B), blk, 0, stream>>>(qkvb, vtb);
        attn_kernel<<<dim3(GPT_T / 64, GPT_H, GPT_B), dim3(128), 0, stream>>>(qkvb, vtb, ybuf);
        gemm_wmma_kernel<GPT_E, GPT_E, false>
            <<<dim3(GPT_E / TN, GPT_M / TM), blk, 0, stream>>>(
            ybuf, proj_w + (size_t)l * GPT_E * GPT_E, proj_b + (size_t)l * GPT_E,
            xbuf, xbuf, nullptr, 0);
        ln_kernel<<<gM, blk, 0, stream>>>(xbuf, ln2_g + l * GPT_E, ln2_b + l * GPT_E, hbuf);
        gemm_wmma_kernel<4 * GPT_E, GPT_E, false>
            <<<dim3(4 * GPT_E / TN, GPT_M / TM), blk, 0, stream>>>(
            hbuf, mlp_w1 + (size_t)l * GPT_E * 4 * GPT_E, mlp_b1 + (size_t)l * 4 * GPT_E,
            nullptr, nullptr, mlph, 1);
        gemm_wmma_kernel<GPT_E, 4 * GPT_E, false>
            <<<dim3(GPT_E / TN, GPT_M / TM), blk, 0, stream>>>(
            mlph, mlp_w2 + (size_t)l * 4 * GPT_E * GPT_E, mlp_b2 + (size_t)l * GPT_E,
            xbuf, xbuf, nullptr, 0);
    }

    ln_kernel<<<gM, blk, 0, stream>>>(xbuf, lnf_g, lnf_b, hbuf);
    gemm_wmma_kernel<GPT_V, GPT_E, true>
        <<<dim3((GPT_V + TN - 1) / TN, GPT_M / TM), blk, 0, stream>>>(
        hbuf, head_w, nullptr, nullptr, out, nullptr, 0);
}
